// ConditionalGraphDiffusion_19370302505312
// MI455X (gfx1250) — matrix operand order repaired and checked by reading, not compiled, not run on hardware
//
#include <hip/hip_runtime.h>
#include <math.h>

// ---------------------------------------------------------------------------
// CDNA5 (gfx1250) implementation of ConditionalGraphDiffusion forward pass.
//
// Dense GEMMs -> block-tiled v_wmma_f32_16x16x32_bf16:
//   * block = 8 waves, tile 128(M) x 64(N), K stepped by 32
//   * A (128x32) + B (32x64, col-major) staged in LDS as bf16 once per step
//   * each wave: 1 A fragment reused across 4 B fragments -> 4 WMMAs
//   * fragments read from LDS as contiguous 128-bit chunks with k = 16*(lane/16) + s
//     for both A and B (not the ISA table k order)
// Graph ops (segment softmax, scatter aggregation, LN, head-softmax
// cross-attn) -> VALU kernels with LDS block reductions.
// ---------------------------------------------------------------------------

#define HID 256
#define GATH 4
#define XHEADS 8
#define XHD 32

typedef __attribute__((ext_vector_type(16))) __bf16 v16bf;
typedef __attribute__((ext_vector_type(8)))  float  v8f;

union BfFrag { v16bf v; unsigned short u[16]; };

__device__ inline unsigned short f2bf_bits(float f) {
  union { float f; unsigned u; } cv; cv.f = f;
  unsigned u = cv.u;
  unsigned r = u + 0x7FFFu + ((u >> 16) & 1u);   // round-to-nearest-even
  return (unsigned short)(r >> 16);
}

__device__ inline float act_apply(float v, int act) {
  switch (act) {
    case 1: return v > 0.f ? v : 0.f;                                   // relu
    case 2: return v / (1.f + __expf(-v));                              // silu
    case 3: return 0.5f * v * (1.f + erff(v * 0.7071067811865475f));    // gelu (exact)
  }
  return v;
}

// ---------------------------------------------------------------------------
// WMMA GEMM: C[M,Nc] = act(A[M,K] @ W[K,Nc] + bias + resid)
// grid: (ceil(M/128), Nc/64), block: 256 threads (8 waves).
// Requires: M % 16 == 0, Nc % 64 == 0, K % 32 == 0 (true for all calls here).
// ---------------------------------------------------------------------------
__global__ __launch_bounds__(256)
void gemm_bf16_wmma_kernel(const float* __restrict__ A, const float* __restrict__ W,
                           const float* __restrict__ bias, const float* __restrict__ resid,
                           float* __restrict__ C, int M, int Nc, int K, int act)
{
  __shared__ unsigned short As[128 * 32];   // [row][k]  row stride 32 (64B)
  __shared__ unsigned short Bs[64 * 32];    // [col][k]  col stride 32 (64B)

  const int tid    = (int)threadIdx.x;
  const int waveId = tid >> 5;              // 0..7 -> M sub-tile
  const int lane   = tid & 31;
  const int lo     = lane & 15;             // row (A frag) / col (B,C frags)
  const int hi     = (lane >> 4) & 1;       // lane-half selects K sub-block

  const int rowBlock = (int)blockIdx.x * 128;
  const int colBlock = (int)blockIdx.y * 64;

  // staging assignments
  const int sa_row  = tid >> 1;             // 0..127
  const int sa_half = tid & 1;              // K halves 0-15 / 16-31
  int sa_rowG = rowBlock + sa_row;
  if (sa_rowG >= M) sa_rowG = M - 1;        // clamp (results discarded by store guard)
  const float* sa_src0 = A + (size_t)sa_rowG * K + sa_half * 16;

  const int sb_k   = tid >> 3;              // 0..31
  const int sb_grp = tid & 7;               // 8 cols each
  const float* sb_src0 = W + (size_t)sb_k * Nc + colBlock + sb_grp * 8;

  v8f acc[4] = {};

  for (int k0 = 0; k0 < K; k0 += 32) {
    // ---- stage A tile (f32 -> bf16) ----
    {
      const float* src = sa_src0 + k0;
      if (k0 + 32 < K) __builtin_prefetch(src + 32, 0, 0);     // global_prefetch_b8
      unsigned short tmp[16];
      #pragma unroll
      for (int j = 0; j < 16; ++j) tmp[j] = f2bf_bits(src[j]);
      __builtin_memcpy(&As[sa_row * 32 + sa_half * 16], tmp, 32);
    }
    // ---- stage B tile (f32 -> bf16, transpose to col-major) ----
    {
      const float* src = sb_src0 + (size_t)k0 * Nc;
      if (k0 + 32 < K) __builtin_prefetch(src + (size_t)32 * Nc, 0, 0);
      float t[8];
      #pragma unroll
      for (int j = 0; j < 8; ++j) t[j] = src[j];
      #pragma unroll
      for (int j = 0; j < 8; ++j)
        Bs[(sb_grp * 8 + j) * 32 + sb_k] = f2bf_bits(t[j]);
    }
    __syncthreads();

    // ---- A fragment: lanes 0-15 take K 0..15, lanes 16-31 take K 16..31
    BfFrag a;
    {
      const unsigned short* ap = &As[(waveId * 16 + lo) * 32];
      __builtin_memcpy(&a.u[0], ap + hi * 16, 32);
    }
    // ---- 4 column tiles: B fragment is 32 contiguous bytes in col-major Bs ----
    #pragma unroll
    for (int c = 0; c < 4; ++c) {
      BfFrag b;
      __builtin_memcpy(&b.u[0], &Bs[(c * 16 + lo) * 32 + hi * 16], 32);
      acc[c] = __builtin_amdgcn_wmma_f32_16x16x32_bf16(false, a.v, false, b.v,
                                                       (short)0, acc[c], false, false);
    }
    __syncthreads();
  }

  // ---- epilogue: bias + residual + activation, guarded for ragged M ----
  #pragma unroll
  for (int c = 0; c < 4; ++c) {
    int col = colBlock + c * 16 + lo;
    float bv = bias ? bias[col] : 0.f;
    #pragma unroll
    for (int v = 0; v < 8; ++v) {
      int row = rowBlock + waveId * 16 + v + hi * 8;   // C layout: VGPR v, lane-half -> M+8
      if (row < M) {
        size_t idx = (size_t)row * Nc + col;
        float val = acc[c][v] + bv;
        if (resid) val += resid[idx];
        C[idx] = act_apply(val, act);
      }
    }
  }
}

// ---------------------------------------------------------------------------
// Elementwise / graph kernels
// ---------------------------------------------------------------------------
__device__ inline float block_reduce_sum256(float v, float* sbuf) {
  int t = threadIdx.x;
  sbuf[t] = v;
  __syncthreads();
  #pragma unroll
  for (int s2 = 128; s2 > 0; s2 >>= 1) {
    if (t < s2) sbuf[t] += sbuf[t + s2];
    __syncthreads();
  }
  float r = sbuf[0];
  __syncthreads();
  return r;
}

__device__ inline void atomicMaxF(float* addr, float val) {
  unsigned* ua = (unsigned*)addr;
  unsigned old = *ua;
  for (;;) {
    float f = __uint_as_float(old);
    if (f >= val) break;
    unsigned assumed = old;
    old = atomicCAS(ua, assumed, __float_as_uint(val));
    if (old == assumed) break;
  }
}

__global__ void fill_kernel(float* p, float v, size_t n) {
  size_t gid = (size_t)blockIdx.x * blockDim.x + threadIdx.x;
  if (gid < n) p[gid] = v;
}

// h = x @ atom_proj + pos @ coord_proj (+ both biases); K = 7 and 3 (too small for WMMA)
__global__ void init_h_kernel(const float* __restrict__ x, const float* __restrict__ pos,
                              const float* __restrict__ aw, const float* __restrict__ ab,
                              const float* __restrict__ cw, const float* __restrict__ cb,
                              float* __restrict__ h, int n) {
  int gid = blockIdx.x * blockDim.x + threadIdx.x;
  if (gid >= n * HID) return;
  int node = gid >> 8, d = gid & 255;
  float s = ab[d] + cb[d];
  const float* xp = x + (size_t)node * 7;
  #pragma unroll
  for (int i = 0; i < 7; ++i) s += xp[i] * aw[i * HID + d];
  const float* pp = pos + (size_t)node * 3;
  #pragma unroll
  for (int i = 0; i < 3; ++i) s += pp[i] * cw[i * HID + d];
  h[gid] = s;
}

// silu(sinusoidal time embedding) -> ste[B,256]
__global__ void time_embed_kernel(const int* __restrict__ t, float* __restrict__ ste, int Bn) {
  int gid = blockIdx.x * blockDim.x + threadIdx.x;
  if (gid >= Bn * HID) return;
  int b = gid >> 8, j = gid & 255;
  int k2 = j & 127;
  float freq = __expf(-(float)k2 * (9.210340371976184f / 127.f));  // ln(1e4)/(half-1)
  float arg = (float)t[b] * freq;
  float v = (j < 128) ? __sinf(arg) : __cosf(arg);
  ste[gid] = v / (1.f + __expf(-v));
}

// a_src/a_dst[n,h] = sum_d xh[n,h,d] * att_{src,dst}[h,d]
__global__ void gat_scores_kernel(const float* __restrict__ xh,
                                  const float* __restrict__ att_src, const float* __restrict__ att_dst,
                                  float* __restrict__ a_src, float* __restrict__ a_dst, int n) {
  int gid = blockIdx.x * blockDim.x + threadIdx.x;
  if (gid >= n * GATH) return;
  int node = gid >> 2, hh = gid & 3;
  const float* xp = xh + (size_t)node * (GATH * HID) + hh * HID;
  const float* as = att_src + hh * HID;
  const float* ad = att_dst + hh * HID;
  float s1 = 0.f, s2 = 0.f;
  for (int d = 0; d < HID; ++d) { float v = xp[d]; s1 += v * as[d]; s2 += v * ad[d]; }
  a_src[gid] = s1; a_dst[gid] = s2;
}

// pass 1: alpha = leaky_relu(a_src[src]+a_dst[dst]); segment max over dst
__global__ void gat_alpha_kernel(const int* __restrict__ ei,
                                 const float* __restrict__ a_src, const float* __restrict__ a_dst,
                                 float* __restrict__ alpha, float* __restrict__ segmax, int E_, int n) {
  int gid = blockIdx.x * blockDim.x + threadIdx.x;
  int tot = (E_ + n) * GATH;
  if (gid >= tot) return;
  int e = gid >> 2, hh = gid & 3;
  int s = (e < E_) ? ei[e]      : (e - E_);     // self-loop tail
  int d = (e < E_) ? ei[E_ + e] : (e - E_);
  float a = a_src[s * GATH + hh] + a_dst[d * GATH + hh];
  a = a > 0.f ? a : 0.2f * a;
  alpha[gid] = a;
  atomicMaxF(&segmax[d * GATH + hh], a);
}

// pass 2: ex = exp(alpha - segmax[dst]); segment sum
__global__ void gat_expsum_kernel(const int* __restrict__ ei,
                                  float* __restrict__ alpha, const float* __restrict__ segmax,
                                  float* __restrict__ segsum, int E_, int n) {
  int gid = blockIdx.x * blockDim.x + threadIdx.x;
  int tot = (E_ + n) * GATH;
  if (gid >= tot) return;
  int e = gid >> 2, hh = gid & 3;
  int d = (e < E_) ? ei[E_ + e] : (e - E_);
  float w = __expf(alpha[gid] - segmax[d * GATH + hh]);
  alpha[gid] = w;                                // reuse buffer as ex
  atomicAdd(&segsum[d * GATH + hh], w);
}

// pass 3: acc[dst,h,:] += (ex/denom) * xh[src,h,:]
__global__ void gat_scatter_kernel(const int* __restrict__ ei, const float* __restrict__ ex,
                                   const float* __restrict__ segsum, const float* __restrict__ xh,
                                   float* __restrict__ acc, int E_, int n) {
  size_t gid = (size_t)blockIdx.x * blockDim.x + threadIdx.x;
  size_t tot = (size_t)(E_ + n) * (GATH * HID);
  if (gid >= tot) return;
  int e = (int)(gid >> 10), j = (int)(gid & 1023), hh = j >> 8;
  int s = (e < E_) ? ei[e]      : (e - E_);
  int d = (e < E_) ? ei[E_ + e] : (e - E_);
  float w = ex[e * GATH + hh] / (segsum[d * GATH + hh] + 1e-16f);
  atomicAdd(&acc[(size_t)d * 1024 + j], w * xh[(size_t)s * 1024 + j]);
}

// h = silu(LN(mean_heads(acc) + gat_bias + time_vec[batch]))  -- one block per node
__global__ void gat_finalize_kernel(const float* __restrict__ acc, const float* __restrict__ gbias,
                                    const float* __restrict__ tv, const int* __restrict__ batch,
                                    const float* __restrict__ lng, const float* __restrict__ lnb,
                                    float* __restrict__ h) {
  __shared__ float sbuf[256];
  int node = blockIdx.x, d = threadIdx.x;
  const float* a = acc + (size_t)node * 1024;
  float v = 0.25f * (a[d] + a[256 + d] + a[512 + d] + a[768 + d])
          + gbias[d] + tv[(size_t)batch[node] * HID + d];
  float mu = block_reduce_sum256(v, sbuf) * (1.f / 256.f);
  float c = v - mu;
  float var = block_reduce_sum256(c * c, sbuf) * (1.f / 256.f);
  float y = c * rsqrtf(var + 1e-5f) * lng[d] + lnb[d];
  h[(size_t)node * HID + d] = y / (1.f + __expf(-y));
}

// plain LayerNorm, one block per row of 256
__global__ void ln_rows_kernel(const float* __restrict__ in, const float* __restrict__ g,
                               const float* __restrict__ b, float* __restrict__ out) {
  __shared__ float sbuf[256];
  int row = blockIdx.x, d = threadIdx.x;
  float v = in[(size_t)row * HID + d];
  float mu = block_reduce_sum256(v, sbuf) * (1.f / 256.f);
  float c = v - mu;
  float var = block_reduce_sum256(c * c, sbuf) * (1.f / 256.f);
  out[(size_t)row * HID + d] = c * rsqrtf(var + 1e-5f) * g[d] + b[d];
}

// cross attention core: per node, scores over 8 heads, softmax across heads, ao = w*v
__global__ void xattn_kernel(const float* __restrict__ q, const float* __restrict__ kp,
                             const float* __restrict__ vp, const int* __restrict__ batch,
                             float* __restrict__ ao, int n) {
  int i = blockIdx.x * blockDim.x + threadIdx.x;
  if (i >= n) return;
  int bg = batch[i];
  const float* qi = q  + (size_t)i  * HID;
  const float* ki = kp + (size_t)bg * HID;
  const float* vi = vp + (size_t)bg * HID;
  float sc[XHEADS];
  float mx = -1e30f;
  #pragma unroll
  for (int hh = 0; hh < XHEADS; ++hh) {
    float s = 0.f;
    for (int d = 0; d < XHD; ++d) s += qi[hh * XHD + d] * ki[hh * XHD + d];
    s *= 0.1767766952966369f;                    // 1/sqrt(32)
    sc[hh] = s; mx = fmaxf(mx, s);
  }
  float den = 0.f;
  #pragma unroll
  for (int hh = 0; hh < XHEADS; ++hh) { sc[hh] = __expf(sc[hh] - mx); den += sc[hh]; }
  float inv = 1.f / den;
  float* aoi = ao + (size_t)i * HID;
  #pragma unroll
  for (int hh = 0; hh < XHEADS; ++hh) {
    float w = sc[hh] * inv;
    for (int d = 0; d < XHD; ++d) aoi[hh * XHD + d] = w * vi[hh * XHD + d];
  }
}

// e_in = edge_attr @ edge_proj (K=3)
__global__ void edge_in_kernel(const float* __restrict__ ea, const float* __restrict__ w,
                               const float* __restrict__ b, float* __restrict__ out, int E_) {
  int gid = blockIdx.x * blockDim.x + threadIdx.x;
  if (gid >= E_ * HID) return;
  int e = gid >> 8, d = gid & 255;
  const float* ep = ea + (size_t)e * 3;
  float s = b[d];
  #pragma unroll
  for (int i = 0; i < 3; ++i) s += ep[i] * w[i * HID + d];
  out[gid] = s;
}

// hidden = relu(P1[row] + P2[col] + P3)  (decomposed [h|h|e_in] @ W1)
__global__ void edge_combine_kernel(const int* __restrict__ ei, const float* __restrict__ P1,
                                    const float* __restrict__ P2, const float* __restrict__ P3,
                                    float* __restrict__ hidden, int E_) {
  int gid = blockIdx.x * blockDim.x + threadIdx.x;
  if (gid >= E_ * HID) return;
  int e = gid >> 8, d = gid & 255;
  int r = ei[e], c = ei[E_ + e];
  float v = P1[(size_t)r * HID + d] + P2[(size_t)c * HID + d] + P3[gid];
  hidden[gid] = v > 0.f ? v : 0.f;
}

// final tiny-N projection: out[M,Nout] = A[M,256] @ W[256,Nout] + b   (Nout = 7 or 3)
__global__ void small_out_kernel(const float* __restrict__ A, const float* __restrict__ W,
                                 const float* __restrict__ b, float* __restrict__ out,
                                 int M, int Nout) {
  int gid = blockIdx.x * blockDim.x + threadIdx.x;
  if (gid >= M * Nout) return;
  int m = gid / Nout, j = gid - m * Nout;
  const float* a = A + (size_t)m * HID;
  float s = b[j];
  for (int k2 = 0; k2 < HID; ++k2) s += a[k2] * W[k2 * Nout + j];
  out[gid] = s;
}

// ---------------------------------------------------------------------------
// Host orchestration
// ---------------------------------------------------------------------------
static void launch_gemm(hipStream_t s, const float* A, const float* W, const float* bias,
                        const float* resid, float* C, int M, int Nc, int K, int act) {
  int tilesM  = M / 16;
  int blocksM = (tilesM + 7) / 8;     // 8 wave-tiles (128 rows) per block
  int blocksN = Nc / 64;              // 64 cols per block
  gemm_bf16_wmma_kernel<<<dim3(blocksM, blocksN), 256, 0, s>>>(A, W, bias, resid, C, M, Nc, K, act);
}

extern "C" void kernel_launch(void* const* d_in, const int* in_sizes, int n_in,
                              void* d_out, int out_size, void* d_ws, size_t ws_size,
                              hipStream_t stream) {
  (void)in_sizes; (void)n_in; (void)out_size; (void)ws_size;
  const int N = 20000, E = 60000, Bn = 512;

  const float* x         = (const float*)d_in[0];   // [N,7]
  const float* pos       = (const float*)d_in[1];   // [N,3]
  const float* edge_attr = (const float*)d_in[2];   // [E,3]
  const int*   ei        = (const int*)  d_in[3];   // [2,E] row-major: src=ei, dst=ei+E
  const int*   batch     = (const int*)  d_in[4];   // [N]
  const int*   timestep  = (const int*)  d_in[5];   // [B]
  const float* prot      = (const float*)d_in[6];   // [B,1280]
  const float* P         = (const float*)d_in[7];   // params, pytree-flat (keys sorted)

  size_t off = 0;
  auto take = [&](size_t cnt) -> const float* { const float* p = P + off; off += cnt; return p; };

  // alphabetical pytree order: atom_out, atom_proj, blocks, coord_out, coord_proj,
  //                            cross, edge_out, edge_proj ; within _lin: b then w
  const float* ao_l1b = take(256);  const float* ao_l1w = take(256 * 256);
  const float* ao_l2b = take(7);    const float* ao_l2w = take(256 * 7);
  const float* ap_b   = take(256);  const float* ap_w   = take(7 * 256);
  struct Blk { const float *att_dst, *att_src, *gat_bias, *gat_w, *ln_b, *ln_g, *time_b, *time_w; } blk[6];
  for (int i = 0; i < 6; ++i) {
    blk[i].att_dst  = take(4 * 256);   blk[i].att_src = take(4 * 256);
    blk[i].gat_bias = take(256);       blk[i].gat_w   = take(256 * 1024);
    blk[i].ln_b     = take(256);       blk[i].ln_g    = take(256);
    blk[i].time_b   = take(256);       blk[i].time_w  = take(256 * 256);
  }
  const float* co_l1b = take(256);  const float* co_l1w = take(256 * 256);
  const float* co_l2b = take(3);    const float* co_l2w = take(256 * 3);
  const float* cp_b   = take(256);  const float* cp_w   = take(3 * 256);
  struct Crs { const float *ff1b, *ff1w, *ff2b, *ff2w, *kb, *kw, *ln1b, *ln1g,
                           *ln2b, *ln2g, *ob, *ow, *qb, *qw, *vb, *vw; } crs[3];
  for (int i = 0; i < 3; ++i) {
    crs[i].ff1b = take(1024); crs[i].ff1w = take(256 * 1024);
    crs[i].ff2b = take(256);  crs[i].ff2w = take(1024 * 256);
    crs[i].kb   = take(256);  crs[i].kw   = take(1280 * 256);
    crs[i].ln1b = take(256);  crs[i].ln1g = take(256);
    crs[i].ln2b = take(256);  crs[i].ln2g = take(256);
    crs[i].ob   = take(256);  crs[i].ow   = take(256 * 256);
    crs[i].qb   = take(256);  crs[i].qw   = take(256 * 256);
    crs[i].vb   = take(256);  crs[i].vw   = take(1280 * 256);
  }
  const float* eo_l1b = take(256);  const float* eo_l1w = take(768 * 256);
  const float* eo_l2b = take(3);    const float* eo_l2w = take(256 * 3);
  const float* ep_b   = take(256);  const float* ep_w   = take(3 * 256);

  // workspace carve-out (~229 MB of fp32)
  float* ws = (float*)d_ws;
  size_t cur = 0;
  auto carve = [&](size_t cnt) -> float* { float* p = ws + cur; cur += cnt; return p; };
  float* bufH   = carve((size_t)N * 256);    // node features h
  float* bufA   = carve((size_t)N * 1024);   // xh / ff1 / edge e_in / edge hidden
  float* bufB   = carve((size_t)N * 1024);   // GAT accumulation / edge P3
  float* bufT1  = carve((size_t)N * 256);
  float* bufT2  = carve((size_t)N * 256);
  float* ste    = carve((size_t)Bn * 256);   // silu(time_emb)
  float* tv     = carve((size_t)Bn * 256);   // per-block time vector
  float* kp     = carve((size_t)Bn * 256);   // per-graph K projection
  float* vp     = carve((size_t)Bn * 256);   // per-graph V projection
  float* a_src  = carve((size_t)N * 4);
  float* a_dst  = carve((size_t)N * 4);
  float* segmax = carve((size_t)N * 4);
  float* segsum = carve((size_t)N * 4);
  float* alphab = carve((size_t)(E + N) * 4);

  float* out = (float*)d_out;                // [N*7 | N*3 | E*3]
  const int T256 = 256;
  auto blocks_for = [](size_t n) -> int { return (int)((n + 255) / 256); };

  // ---- input projections + time embedding ----
  init_h_kernel<<<blocks_for((size_t)N * 256), T256, 0, stream>>>(x, pos, ap_w, ap_b, cp_w, cp_b, bufH, N);
  time_embed_kernel<<<blocks_for((size_t)Bn * 256), T256, 0, stream>>>(timestep, ste, Bn);

  int cross_idx = 0;
  for (int l = 0; l < 6; ++l) {
    const Blk& bp = blk[l];
    // GAT
    launch_gemm(stream, bufH, bp.gat_w, nullptr, nullptr, bufA, N, 1024, 256, 0);   // xh
    gat_scores_kernel<<<blocks_for((size_t)N * 4), T256, 0, stream>>>(bufA, bp.att_src, bp.att_dst, a_src, a_dst, N);
    fill_kernel<<<blocks_for((size_t)N * 4), T256, 0, stream>>>(segmax, -3e38f, (size_t)N * 4);
    fill_kernel<<<blocks_for((size_t)N * 4), T256, 0, stream>>>(segsum, 0.f, (size_t)N * 4);
    fill_kernel<<<blocks_for((size_t)N * 1024), T256, 0, stream>>>(bufB, 0.f, (size_t)N * 1024);
    gat_alpha_kernel<<<blocks_for((size_t)(E + N) * 4), T256, 0, stream>>>(ei, a_src, a_dst, alphab, segmax, E, N);
    gat_expsum_kernel<<<blocks_for((size_t)(E + N) * 4), T256, 0, stream>>>(ei, alphab, segmax, segsum, E, N);
    gat_scatter_kernel<<<blocks_for((size_t)(E + N) * 1024), T256, 0, stream>>>(ei, alphab, segsum, bufA, bufB, E, N);
    // time conditioning (per graph) + LN + silu
    launch_gemm(stream, ste, bp.time_w, bp.time_b, nullptr, tv, Bn, 256, 256, 0);
    gat_finalize_kernel<<<N, T256, 0, stream>>>(bufB, bp.gat_bias, tv, batch, bp.ln_g, bp.ln_b, bufH);

    if ((l & 1) && cross_idx < 3) {
      const Crs& cp2 = crs[cross_idx++];
      launch_gemm(stream, prot, cp2.kw, cp2.kb, nullptr, kp, Bn, 256, 1280, 0);     // K per graph
      launch_gemm(stream, prot, cp2.vw, cp2.vb, nullptr, vp, Bn, 256, 1280, 0);     // V per graph
      ln_rows_kernel<<<N, T256, 0, stream>>>(bufH, cp2.ln1g, cp2.ln1b, bufT1);      // hn
      launch_gemm(stream, bufT1, cp2.qw, cp2.qb, nullptr, bufT2, N, 256, 256, 0);   // q
      xattn_kernel<<<blocks_for((size_t)N), T256, 0, stream>>>(bufT2, kp, vp, batch, bufT1, N); // ao
      launch_gemm(stream, bufT1, cp2.ow, cp2.ob, bufH, bufH, N, 256, 256, 0);       // h += ao@Wo
      ln_rows_kernel<<<N, T256, 0, stream>>>(bufH, cp2.ln2g, cp2.ln2b, bufT1);
      launch_gemm(stream, bufT1, cp2.ff1w, cp2.ff1b, nullptr, bufA, N, 1024, 256, 3);  // gelu
      launch_gemm(stream, bufA, cp2.ff2w, cp2.ff2b, bufH, bufH, N, 256, 1024, 0);      // h += ff
    }
  }

  // ---- output heads ----
  launch_gemm(stream, bufH, ao_l1w, ao_l1b, nullptr, bufT1, N, 256, 256, 1);        // relu
  small_out_kernel<<<blocks_for((size_t)N * 7), T256, 0, stream>>>(bufT1, ao_l2w, ao_l2b, out, N, 7);
  launch_gemm(stream, bufH, co_l1w, co_l1b, nullptr, bufT1, N, 256, 256, 1);
  small_out_kernel<<<blocks_for((size_t)N * 3), T256, 0, stream>>>(bufT1, co_l2w, co_l2b, out + (size_t)N * 7, N, 3);

  // edge head: [h[row] | h[col] | e_in] @ W1 decomposed into 3 GEMMs + gather combine
  edge_in_kernel<<<blocks_for((size_t)E * 256), T256, 0, stream>>>(edge_attr, ep_w, ep_b, bufA, E);
  launch_gemm(stream, bufH, eo_l1w,                     nullptr, nullptr, bufT1, N, 256, 256, 0); // P1
  launch_gemm(stream, bufH, eo_l1w + (size_t)256 * 256, nullptr, nullptr, bufT2, N, 256, 256, 0); // P2
  launch_gemm(stream, bufA, eo_l1w + (size_t)512 * 256, eo_l1b,  nullptr, bufB,  E, 256, 256, 0); // P3 (+bias)
  edge_combine_kernel<<<blocks_for((size_t)E * 256), T256, 0, stream>>>(ei, bufT1, bufT2, bufB, bufA, E);
  small_out_kernel<<<blocks_for((size_t)E * 3), T256, 0, stream>>>(bufA, eo_l2w, eo_l2b, out + (size_t)N * 10, E, 3);
}
